// HyperSE_43834436223783
// MI455X (gfx1250) — matrix operand order repaired and checked by reading, not compiled, not run on hardware
//
#include <hip/hip_runtime.h>
#include <hip/hip_bf16.h>

#define N_NODES 10000
#define E_EDGES 320000
#define F_INF   512
#define HID     256
#define D_HYP   2
#define MIN_NORMF 1e-15f
#define MIN_SCALE 0.666f   // HEIGHT * MAX_SIZE/(HEIGHT+1) = 2*0.999/3
#define MAX_SCALE 0.999f

#define SA 40   // LDS A row stride in bf16 (80B: 16B-aligned rows, good bank spread)
#define SB 40   // LDS B column stride in bf16 (transposed tile)

typedef __attribute__((ext_vector_type(16))) __bf16 v16bf;
typedef __attribute__((ext_vector_type(8)))  float  v8f;

union Frag16 { v16bf v; __bf16 e[16]; float4 q[2]; };
union FragC  { v8f   v; float  e[8];  };

__device__ __forceinline__ unsigned int pack_bf2(float a, float b) {
    union { __bf16 h[2]; unsigned int u; } p;
    p.h[0] = (__bf16)a; p.h[1] = (__bf16)b;
    return p.u;
}

// ---------------- utility kernels ----------------
__global__ void zero_f32(float* __restrict__ p, int n) {
    int i = blockIdx.x * blockDim.x + threadIdx.x;
    if (i < n) p[i] = 0.0f;
}

__global__ void deg_accum(const int* __restrict__ dst, float* __restrict__ deg) {
    int e = blockIdx.x * blockDim.x + threadIdx.x;
    if (e < E_EDGES) atomicAdd(&deg[dst[e]], 1.0f);
}

__global__ void deg_clamp(float* __restrict__ deg) {
    int i = blockIdx.x * blockDim.x + threadIdx.x;
    if (i < N_NODES) deg[i] = fmaxf(deg[i], 1.0f);
}

__global__ void edge_norm(const int* __restrict__ src, const int* __restrict__ dst,
                          const float* __restrict__ deg, float* __restrict__ enorm) {
    int e = blockIdx.x * blockDim.x + threadIdx.x;
    if (e < E_EDGES) enorm[e] = __frsqrt_rn(deg[src[e]] * deg[dst[e]]);
}

// ---------------- GEMM1: H0 = feature @ W1 + b1 via bf16 WMMA ----------------
// Block = 256 threads (8 waves); 16-row strip x 256 cols; K in steps of 32.
// LDS layouts chosen so each WMMA fragment is two aligned ds_load_b128s.
__global__ __launch_bounds__(256) void gemm1_wmma(const float* __restrict__ feat,
                                                  const float* __restrict__ W1,
                                                  const float* __restrict__ b1,
                                                  float* __restrict__ H0) {
    __shared__ __bf16 ldsA[16 * SA];      // A tile: [row m][k], row stride SA
    __shared__ __bf16 ldsB[HID * SB];     // B tile TRANSPOSED: [col c][k], stride SB

    const int tid  = threadIdx.x;
    const int wave = tid >> 5;
    const int lane = tid & 31;
    const int half = lane >> 4;           // 0/1 : lanes 0-15 / 16-31
    const int mn   = lane & 15;           // row (A) / col (B,C) within tile
    const int row_base = blockIdx.x * 16;

    unsigned int* ldsA32 = (unsigned int*)ldsA;
    unsigned int* ldsB32 = (unsigned int*)ldsB;

    FragC acc0; FragC acc1;
    #pragma unroll
    for (int r = 0; r < 8; ++r) { acc0.e[r] = 0.0f; acc1.e[r] = 0.0f; }

    for (int ks = 0; ks < F_INF / 32; ++ks) {
        const int kb = ks * 32;
        __syncthreads();
        // ---- cooperative A fill: 16x32 f32 -> packed bf16, one b32 store/thread
        {
            int idx = tid * 2;                 // 512 elems, 2 per thread
            int m = idx >> 5, k = idx & 31;    // k even
            const float2 f = *(const float2*)&feat[(size_t)(row_base + m) * F_INF + kb + k];
            ldsA32[(m * SA + k) >> 1] = pack_bf2(f.x, f.y);
        }
        // ---- cooperative B fill (transposed): column c = tid, pack k-pairs
        #pragma unroll 4
        for (int j = 0; j < 32; j += 2) {
            float f0 = W1[(size_t)(kb + j) * HID + tid];
            float f1 = W1[(size_t)(kb + j + 1) * HID + tid];
            ldsB32[(tid * SB + j) >> 1] = pack_bf2(f0, f1);
        }
        __syncthreads();

        // ---- A fragment: e[0..7] = k half*8.., e[8..15] = k 16+half*8.. (row mn)
        Frag16 afrag;
        afrag.q[0] = *(const float4*)&ldsA[mn * SA + half * 8];
        afrag.q[1] = *(const float4*)&ldsA[mn * SA + 16 + half * 8];

        // ---- two column tiles per wave
        #pragma unroll
        for (int t = 0; t < 2; ++t) {
            const int cb = wave * 32 + t * 16;
            // B fragment: e[0..7] = k half*8.., e[8..15] = k 16+half*8.. at column cb+mn
            Frag16 bfrag;
            bfrag.q[0] = *(const float4*)&ldsB[(cb + mn) * SB + half * 8];
            bfrag.q[1] = *(const float4*)&ldsB[(cb + mn) * SB + 16 + half * 8];
            if (t == 0)
                acc0.v = __builtin_amdgcn_wmma_f32_16x16x32_bf16(
                    false, afrag.v, false, bfrag.v, (short)0, acc0.v, false, false);
            else
                acc1.v = __builtin_amdgcn_wmma_f32_16x16x32_bf16(
                    false, afrag.v, false, bfrag.v, (short)0, acc1.v, false, false);
        }
    }

    // C/D layout: lane L, reg r -> row = r + 8*(L/16), col = L%16
    #pragma unroll
    for (int r = 0; r < 8; ++r) {
        int row = row_base + r + half * 8;
        int c0  = wave * 32 + mn;
        H0[(size_t)row * HID + c0]      = acc0.e[r] + b1[c0];
        H0[(size_t)row * HID + c0 + 16] = acc1.e[r] + b1[c0 + 16];
    }
}

// ---------------- edge scatter, HID channels ----------------
// 4 edges per block; 64 threads/edge; 4 channels/thread via float4 gather.
__global__ __launch_bounds__(256) void agg_hid(const int* __restrict__ src,
                                               const int* __restrict__ dst,
                                               const float* __restrict__ enorm,
                                               const float* __restrict__ H0,
                                               float* __restrict__ out) {
    int e = blockIdx.x * 4 + (threadIdx.x >> 6);
    if (e >= E_EDGES) return;
    int c = (threadIdx.x & 63) * 4;
    int s = src[e], d = dst[e];
    float w = enorm[e];
    const float4 v = *(const float4*)&H0[(size_t)s * HID + c];
    float* o = &out[(size_t)d * HID + c];
    atomicAdd(o + 0, v.x * w);
    atomicAdd(o + 1, v.y * w);
    atomicAdd(o + 2, v.z * w);
    atomicAdd(o + 3, v.w * w);
}

// ---------------- GEMM2 with fused ReLU: u = relu(agg1) @ W2 + b2 ----------------
__global__ __launch_bounds__(256) void gemm2_relu(const float* __restrict__ agg1,
                                                  const float* __restrict__ W2,
                                                  const float* __restrict__ b2,
                                                  float* __restrict__ u) {
    int gwave = (blockIdx.x * blockDim.x + threadIdx.x) >> 5;
    int lane  = threadIdx.x & 31;
    if (gwave >= N_NODES) return;
    const float* row = &agg1[(size_t)gwave * HID];
    float s0 = 0.0f, s1 = 0.0f;
    #pragma unroll
    for (int j = 0; j < HID / 32; ++j) {
        int c = j * 32 + lane;
        float h = fmaxf(row[c], 0.0f);
        s0 = fmaf(h, W2[c * 2 + 0], s0);
        s1 = fmaf(h, W2[c * 2 + 1], s1);
    }
    #pragma unroll
    for (int off = 16; off > 0; off >>= 1) {
        s0 += __shfl_down(s0, off, 32);
        s1 += __shfl_down(s1, off, 32);
    }
    if (lane == 0) {
        u[gwave * 2 + 0] = s0 + b2[0];
        u[gwave * 2 + 1] = s1 + b2[1];
    }
}

// ---------------- edge scatter, D_HYP=2 channels ----------------
__global__ void agg_d(const int* __restrict__ src, const int* __restrict__ dst,
                      const float* __restrict__ enorm, const float* __restrict__ u,
                      float* __restrict__ out) {
    int e = blockIdx.x * blockDim.x + threadIdx.x;
    if (e >= E_EDGES) return;
    int s = src[e], d = dst[e];
    float w = enorm[e];
    atomicAdd(&out[d * 2 + 0], u[s * 2 + 0] * w);
    atomicAdd(&out[d * 2 + 1], u[s * 2 + 1] * w);
}

// ---------------- hyperbolic lift + Poincare projection ----------------
__global__ void hyp_out(const float* __restrict__ u, const float* __restrict__ scale,
                        float* __restrict__ out) {
    int i = blockIdx.x * blockDim.x + threadIdx.x;
    if (i >= N_NODES) return;
    float ux = u[i * 2 + 0], uy = u[i * 2 + 1];
    float un = fmaxf(sqrtf(ux * ux + uy * uy), MIN_NORMF);
    float ch = coshf(un), sh = sinhf(un);
    float inv = sh / (un * (1.0f + ch));
    float px = ux * inv, py = uy * inv;
    float pn = fmaxf(sqrtf(px * px + py * py), 1e-12f);
    float s  = fminf(fmaxf(scale[0], MIN_SCALE), MAX_SCALE);
    px = px / pn * s;
    py = py / pn * s;
    float nrm = fmaxf(sqrtf(px * px + py * py), MIN_NORMF);
    const float maxnorm = 1.0f - MIN_NORMF;
    if (nrm > maxnorm) { px = px / nrm * maxnorm; py = py / nrm * maxnorm; }
    out[i * 2 + 0] = px;
    out[i * 2 + 1] = py;
}

extern "C" void kernel_launch(void* const* d_in, const int* in_sizes, int n_in,
                              void* d_out, int out_size, void* d_ws, size_t ws_size,
                              hipStream_t stream) {
    const float* feat  = (const float*)d_in[0];
    const int*   eidx  = (const int*)d_in[1];
    const float* W1    = (const float*)d_in[2];
    const float* b1    = (const float*)d_in[3];
    const float* W2    = (const float*)d_in[4];
    const float* b2    = (const float*)d_in[5];
    const float* scale = (const float*)d_in[6];
    const int* src = eidx;
    const int* dst = eidx + E_EDGES;

    float* ws    = (float*)d_ws;
    float* deg   = ws;                                    // N
    float* enorm = deg + N_NODES;                         // E
    float* H0    = enorm + E_EDGES;                       // N*HID
    float* agg1  = H0 + (size_t)N_NODES * HID;            // N*HID
    float* u     = agg1 + (size_t)N_NODES * HID;          // N*2
    float* agg2  = u + 2 * N_NODES;                       // N*2

    zero_f32<<<(N_NODES + 255) / 256, 256, 0, stream>>>(deg, N_NODES);
    zero_f32<<<(N_NODES * HID + 255) / 256, 256, 0, stream>>>(agg1, N_NODES * HID);
    zero_f32<<<(2 * N_NODES + 255) / 256, 256, 0, stream>>>(agg2, 2 * N_NODES);

    deg_accum<<<(E_EDGES + 255) / 256, 256, 0, stream>>>(dst, deg);
    deg_clamp<<<(N_NODES + 255) / 256, 256, 0, stream>>>(deg);
    edge_norm<<<(E_EDGES + 255) / 256, 256, 0, stream>>>(src, dst, deg, enorm);

    gemm1_wmma<<<N_NODES / 16, 256, 0, stream>>>(feat, W1, b1, H0);
    agg_hid<<<E_EDGES / 4, 256, 0, stream>>>(src, dst, enorm, H0, agg1);
    gemm2_relu<<<(N_NODES + 7) / 8, 256, 0, stream>>>(agg1, W2, b2, u);
    agg_d<<<(E_EDGES + 255) / 256, 256, 0, stream>>>(src, dst, enorm, u, agg2);
    hyp_out<<<(N_NODES + 255) / 256, 256, 0, stream>>>(agg2, scale, (float*)d_out);
}
